// ModularBrainGraphTransformer_77498389889096
// MI455X (gfx1250) — matrix operand order repaired and checked by reading, not compiled, not run on hardware
//
#include <hip/hip_runtime.h>

// ---------------------------------------------------------------------------
// Problem constants (from the reference)
// ---------------------------------------------------------------------------
constexpr int kM   = 7;            // modules
constexpr int kNM  = 1024;         // nodes / module
constexpr int kEM  = 32768;        // edges / module (without self loops)
constexpr int kET  = kEM + kNM;    // edges + self loops
constexpr int kN   = kM * kNM;     // total nodes
constexpr int kE   = kM * kEM;     // total edges
constexpr int kF   = 128;          // input node features
constexpr int kD   = 64;           // hidden channels
constexpr int kH   = 4;            // heads
constexpr int kHC  = kH * kD;      // 256
constexpr int kQC  = 3 * kD;       // 192 (QKV concat)

typedef __attribute__((ext_vector_type(16))) __bf16 v16bf;
typedef __attribute__((ext_vector_type(8)))  __bf16 v8bf;
typedef __attribute__((ext_vector_type(4)))  __bf16 v4bf;
typedef __attribute__((ext_vector_type(8)))  float  v8f;
typedef __attribute__((ext_vector_type(4)))  float  v4f;

__device__ inline v16bf cat8(v8bf lo, v8bf hi) {
  union { v16bf v; v8bf h[2]; } u;
  u.h[0] = lo; u.h[1] = hi;
  return u.v;
}

// order-preserving float<->uint map for atomicMax on floats
__device__ inline unsigned fenc(float f) {
  unsigned u = __float_as_uint(f);
  return (u & 0x80000000u) ? ~u : (u | 0x80000000u);
}
__device__ inline float fdec(unsigned e) {
  return (e & 0x80000000u) ? __uint_as_float(e ^ 0x80000000u)
                           : __uint_as_float(~e);
}
// local (src,dst) for edge e of module m; e >= kEM are self loops
__device__ inline void edge_sd(const int* __restrict__ ei, int m, int e,
                               int& s, int& d) {
  if (e < kEM) {
    int gi = m * kEM + e;
    s = ei[gi]      - m * kNM;
    d = ei[kE + gi] - m * kNM;
  } else {
    s = d = e - kEM;
  }
}

// ---------------------------------------------------------------------------
// Utility kernels
// ---------------------------------------------------------------------------
__global__ void k_fill_u32(unsigned* __restrict__ p, unsigned v, int n) {
  int i = blockIdx.x * blockDim.x + threadIdx.x;
  if (i < n) p[i] = v;
}

// 4-wide f32 -> bf16 conversion (all our sizes are multiples of 4)
__global__ void k_cvt_bf16_v4(const float* __restrict__ in, __bf16* __restrict__ out, int n4) {
  int i = blockIdx.x * blockDim.x + threadIdx.x;
  if (i >= n4) return;
  v4f x = ((const v4f*)in)[i];
  v4bf y;
#pragma unroll
  for (int j = 0; j < 4; ++j) y[j] = (__bf16)x[j];
  ((v4bf*)out)[i] = y;
}

// Pack a (optionally transposed) f32 weight matrix into the per-lane WMMA-B
// layout: out[m][nt][kt][n][grp][e] with k = kt*32 + (e&7) + 8*grp + 16*(e>>3), col = nt*16+n.
// A B-fragment load then becomes ONE contiguous 32-byte load per lane.
__global__ void k_pack_b(const float* __restrict__ in, __bf16* __restrict__ out,
                         int K, int N, int trans, int total) {
  int tid = blockIdx.x * blockDim.x + threadIdx.x;
  if (tid >= total) return;
  int e    = tid & 15;
  int grp  = (tid >> 4) & 1;
  int n    = (tid >> 5) & 15;
  int rest = tid >> 9;
  int KT = K >> 5, NT = N >> 4;
  int kt = rest % KT; rest /= KT;
  int nt = rest % NT; int m = rest / NT;
  int k   = (kt << 5) + (e & 7) + (grp << 3) + ((e >> 3) << 4);
  int col = (nt << 4) + n;
  float v = trans ? in[((size_t)m * N + col) * K + k]
                  : in[((size_t)m * K + k) * N + col];
  out[tid] = (__bf16)v;
}

// Pack V (with bias) into WMMA-B layout per (module, head):
// vpk[m][head][kt][n][grp][e], k = key index, n = dh column.
__global__ void k_pack_v(const float* __restrict__ qkvf, const float* __restrict__ in_b,
                         __bf16* __restrict__ vpk) {
  int tid = blockIdx.x * blockDim.x + threadIdx.x;
  if (tid >= kM * kH * kNM * 16) return;
  int e = tid & 15, grp = (tid >> 4) & 1, n = (tid >> 5) & 15;
  int kt = (tid >> 9) & 31;
  int head = (tid >> 14) & 3;
  int m = tid >> 16;
  int k = (kt << 5) + (e & 7) + (grp << 3) + ((e >> 3) << 4);
  int c = 128 + head * 16 + n;
  vpk[tid] = (__bf16)(qkvf[((size_t)(m * kNM) + k) * kQC + c] + in_b[m * kQC + c]);
}

// ---------------------------------------------------------------------------
// Generic batched WMMA GEMM: C[m] = A[m](MxK bf16 row-major) * Bpk[m] (packed)
// One wave per 16x16 tile. A fragment = 2 x b128, B fragment = 1 x 32B load.
// ---------------------------------------------------------------------------
__global__ void __launch_bounds__(256)
k_gemm_bf16(const __bf16* __restrict__ A, const __bf16* __restrict__ Bpk,
            float* __restrict__ C, int Md, int Nd, int Kd,
            int lda, int ldc, long long sA, long long sB, long long sC) {
  int mb    = blockIdx.z;
  int wave  = (blockIdx.x * blockDim.x + threadIdx.x) >> 5;   // wave-uniform
  int lane  = threadIdx.x & 31;
  int tilesN = Nd >> 4, KT = Kd >> 5;
  int tiles  = (Md >> 4) * tilesN;
  if (wave >= tiles) return;                                   // whole-wave exit
  int tm = wave / tilesN, tn = wave - tm * tilesN;
  int rc = lane & 15, grp = lane >> 4;
  const __bf16* Ap = A + mb * sA + (size_t)(tm * 16 + rc) * lda + (grp << 3);
  const __bf16* Bp = Bpk + mb * sB + (size_t)tn * KT * 512 + (((rc << 1) | grp) << 4);
  float* Cp = C + mb * sC + (size_t)(tm * 16) * ldc + tn * 16;
  v8f acc = {};
  for (int kt = 0; kt < KT; ++kt) {
    v16bf a = cat8(*(const v8bf*)(Ap + kt * 32),
                   *(const v8bf*)(Ap + kt * 32 + 16));
    v16bf b = *(const v16bf*)(Bp + kt * 512);
    acc = __builtin_amdgcn_wmma_f32_16x16x32_bf16(false, a, false, b,
                                                  (short)0, acc, false, false);
  }
#pragma unroll
  for (int r = 0; r < 8; ++r)
    Cp[(size_t)(r + grp * 8) * ldc + rc] = acc[r];             // C/D layout
}

// ---------------------------------------------------------------------------
// GAT pieces
// ---------------------------------------------------------------------------
__global__ void k_att_coef(const float* __restrict__ h, const float* __restrict__ as_,
                           const float* __restrict__ ad_, float* __restrict__ es,
                           float* __restrict__ ed, int Hh, int Cc) {
  int tid = blockIdx.x * blockDim.x + threadIdx.x;
  if (tid >= kN * Hh) return;
  int head = tid % Hh, n = tid / Hh, m = n / kNM;
  const float* hp = h   + (size_t)n * Hh * Cc + head * Cc;
  const float* ap = as_ + (size_t)(m * Hh + head) * Cc;
  const float* dp = ad_ + (size_t)(m * Hh + head) * Cc;
  float a = 0.f, b = 0.f;
  for (int c = 0; c < Cc; ++c) { a += hp[c] * ap[c]; b += hp[c] * dp[c]; }
  es[tid] = a; ed[tid] = b;
}

__global__ void k_edge_p1(const int* __restrict__ ei, const float* __restrict__ es,
                          const float* __restrict__ ed, float* __restrict__ ev,
                          unsigned* __restrict__ mx, int Hh) {
  int tid = blockIdx.x * blockDim.x + threadIdx.x;
  if (tid >= kM * kET * Hh) return;
  int head = tid % Hh, e = (tid / Hh) % kET, m = tid / (Hh * kET);
  int s, d; edge_sd(ei, m, e, s, d);
  float v = es[(m * kNM + s) * Hh + head] + ed[(m * kNM + d) * Hh + head];
  v = v > 0.f ? v : 0.2f * v;
  ev[tid] = v;
  atomicMax(&mx[(m * kNM + d) * Hh + head], fenc(v));
}

__global__ void k_edge_p2(const int* __restrict__ ei, float* __restrict__ ev,
                          const unsigned* __restrict__ mx, float* __restrict__ den, int Hh) {
  int tid = blockIdx.x * blockDim.x + threadIdx.x;
  if (tid >= kM * kET * Hh) return;
  int head = tid % Hh, e = (tid / Hh) % kET, m = tid / (Hh * kET);
  int s, d; edge_sd(ei, m, e, s, d);
  float ex = __expf(ev[tid] - fdec(mx[(m * kNM + d) * Hh + head]));
  ev[tid] = ex;
  atomicAdd(&den[(m * kNM + d) * Hh + head], ex);
}

// gather from bf16 shadow of h (halves the dominant gather traffic);
// accumulate in f32 via L2 atomics.
__global__ void k_edge_p3(const int* __restrict__ ei, const float* __restrict__ ev,
                          const float* __restrict__ den, const __bf16* __restrict__ hsrc,
                          float* __restrict__ out, int Hh, int Cc) {
  int CC  = Hh * Cc;
  int tid = blockIdx.x * blockDim.x + threadIdx.x;
  if (tid >= kM * kET * CC) return;
  int col = tid % CC, rest = tid / CC;
  int e = rest % kET, m = rest / kET;
  int head = col / Cc;
  int s, d; edge_sd(ei, m, e, s, d);
  float alpha = ev[((size_t)(m * kET) + e) * Hh + head] /
                (den[(m * kNM + d) * Hh + head] + 1e-16f);
  atomicAdd(&out[(size_t)(m * kNM + d) * CC + col],
            alpha * (float)hsrc[(size_t)(m * kNM + s) * CC + col]);
}

__global__ void k_bias_relu_cvt(const float* __restrict__ in, const float* __restrict__ bias,
                                __bf16* __restrict__ out, int Cc) {
  int tid = blockIdx.x * blockDim.x + threadIdx.x;
  if (tid >= kN * Cc) return;
  int c = tid % Cc, m = tid / (kNM * Cc);
  float v = in[tid] + bias[m * Cc + c];
  out[tid] = (__bf16)(v > 0.f ? v : 0.f);
}

__global__ void k_bias_cvt(const float* __restrict__ in, const float* __restrict__ bias,
                           __bf16* __restrict__ out, int Cc) {
  int tid = blockIdx.x * blockDim.x + threadIdx.x;
  if (tid >= kN * Cc) return;
  int c = tid % Cc, m = tid / (kNM * Cc);
  out[tid] = (__bf16)(in[tid] + bias[m * Cc + c]);
}

// ---------------------------------------------------------------------------
// Fused self-attention: one block = (module, head, 16-query tile).
// 8 waves: QK^T via WMMA into LDS (bf16), row softmax in LDS, PV via WMMA.
// Q frag: 1 x b128 (+zero pad half).  K frag: 1 x b128 (+zero pad half).
// P frag: 2 x ds_load_b128.  V frag: 1 contiguous v16bf from packed vpk.
// ---------------------------------------------------------------------------
__global__ void __launch_bounds__(256)
k_attn(const __bf16* __restrict__ qkv, const __bf16* __restrict__ vpk,
       float* __restrict__ out) {
  __shared__ __bf16 sS[16 * 1024];
  __shared__ float  sPart[8 * 16 * 16];
  __shared__ float  sRed[256];
  int m = blockIdx.z, head = blockIdx.y, q0 = blockIdx.x * 16;
  const __bf16* base = qkv + (size_t)m * kNM * kQC;
  const __bf16* vp   = vpk + (size_t)(m * kH + head) * (kNM * 16)
                           + (((threadIdx.x & 31 & 15) << 1) << 4); // refined below
  int t = threadIdx.x, lane = t & 31, wv = t >> 5;
  int rc = lane & 15, grp = lane >> 4;
  vp = vpk + (size_t)(m * kH + head) * (kNM * 16) + (((rc << 1) | grp) << 4);

  v16bf z16 = {};
  v8bf  z8  = {};

  // Q fragment (invariant over key tiles): dh=16 padded to K=32 with zeros
  v16bf aq = cat8(*(const v8bf*)(base + (size_t)(q0 + rc) * kQC + head * 16 + (grp << 3)),
                  z8);

  // --- S = (Q K^T) * 1/sqrt(16) --------------------------------------------
  for (int kt = wv; kt < 64; kt += 8) {            // wave-uniform loop
    int kcol0 = kt * 16;
    v8bf kv = *(const v8bf*)(base + (size_t)(kcol0 + rc) * kQC + 64 + head * 16 + (grp << 3));
    v16bf b  = cat8(kv, z8);                       // elements 8..15 (K>=16) are zero pad
    v8f c = {};
    c = __builtin_amdgcn_wmma_f32_16x16x32_bf16(false, aq, false, b,
                                                (short)0, c, false, false);
#pragma unroll
    for (int r = 0; r < 8; ++r)
      sS[(r + grp * 8) * 1024 + kcol0 + rc] = (__bf16)(c[r] * 0.25f);
  }
  __syncthreads();

  // --- row softmax: 16 threads per row -------------------------------------
  int r2 = t >> 4, s2 = t & 15;
  float mxv = -1e30f;
  for (int jx = s2; jx < 1024; jx += 16) mxv = fmaxf(mxv, (float)sS[r2 * 1024 + jx]);
  sRed[t] = mxv; __syncthreads();
  if (s2 == 0) {
    float v = sRed[r2 * 16];
    for (int i2 = 1; i2 < 16; ++i2) v = fmaxf(v, sRed[r2 * 16 + i2]);
    sRed[r2 * 16] = v;
  }
  __syncthreads();
  mxv = sRed[r2 * 16];
  __syncthreads();
  float sum = 0.f;
  for (int jx = s2; jx < 1024; jx += 16) {
    float e2 = __expf((float)sS[r2 * 1024 + jx] - mxv);
    sS[r2 * 1024 + jx] = (__bf16)e2;
    sum += e2;
  }
  sRed[t] = sum; __syncthreads();
  if (s2 == 0) {
    float v = 0.f;
    for (int i2 = 0; i2 < 16; ++i2) v += sRed[r2 * 16 + i2];
    sRed[r2 * 16] = v;
  }
  __syncthreads();
  float inv = 1.f / sRed[r2 * 16];
  for (int jx = s2; jx < 1024; jx += 16)
    sS[r2 * 1024 + jx] = (__bf16)((float)sS[r2 * 1024 + jx] * inv);
  __syncthreads();

  // --- O = P V ; each wave owns 4 K-chunks of 32, partials reduced in LDS --
  v8f acc = {};
  for (int kc = 0; kc < 4; ++kc) {
    int kt  = wv * 4 + kc;
    int kk0 = kt * 32;
    v16bf a = cat8(*(const v8bf*)&sS[rc * 1024 + kk0 + (grp << 3)],
                   *(const v8bf*)&sS[rc * 1024 + kk0 + 16 + (grp << 3)]);
    v16bf b = *(const v16bf*)(vp + kt * 512);
    acc = __builtin_amdgcn_wmma_f32_16x16x32_bf16(false, a, false, b,
                                                  (short)0, acc, false, false);
  }
#pragma unroll
  for (int r = 0; r < 8; ++r)
    sPart[wv * 256 + (r + grp * 8) * 16 + rc] = acc[r];
  __syncthreads();
  {
    int qr = t >> 4, nn = t & 15;
    float v = 0.f;
    for (int w2 = 0; w2 < 8; ++w2) v += sPart[w2 * 256 + qr * 16 + nn];
    out[(size_t)(m * kNM + q0 + qr) * kD + head * 16 + nn] = v;
  }
}

// ---------------------------------------------------------------------------
// reps[m] = mean_rows(mha[m]) + sa_out_b[m]   (bias folded through the mean)
// ---------------------------------------------------------------------------
__global__ void k_rowmean(const float* __restrict__ mha, const float* __restrict__ outb,
                          float* __restrict__ reps) {
  __shared__ float red[256];
  int m = blockIdx.x, t = threadIdx.x, c = t & 63, seg = t >> 6;
  float sum = 0.f;
  for (int r = seg; r < kNM; r += 4) sum += mha[((size_t)m * kNM + r) * kD + c];
  red[t] = sum; __syncthreads();
  if (seg == 0)
    reps[m * kD + c] = (red[c] + red[64 + c] + red[128 + c] + red[192 + c]) *
                       (1.f / kNM) + outb[m * kD + c];
}

// ---------------------------------------------------------------------------
// Tiny head: 7-token cross-MHA + fusion softmax + fc + classifier. One block.
// ---------------------------------------------------------------------------
__global__ void k_head(const float* __restrict__ reps,
                       const float* __restrict__ ca_in_w, const float* __restrict__ ca_in_b,
                       const float* __restrict__ ca_out_w, const float* __restrict__ ca_out_b,
                       const float* __restrict__ fusion_w,
                       const float* __restrict__ fc_w, const float* __restrict__ fc_b,
                       const float* __restrict__ c1_w, const float* __restrict__ c1_b,
                       const float* __restrict__ c2_w, const float* __restrict__ c2_b,
                       float* __restrict__ outp) {
  __shared__ float sR[7 * 64], sQ[7 * kQC], sO[7 * 64], sC[7 * 64];
  __shared__ float sW[7], sF[64], sHb[32];
  int t = threadIdx.x;
  for (int i = t; i < 7 * 64; i += 64) sR[i] = reps[i];
  __syncthreads();
  for (int idx = t; idx < 7 * kQC; idx += 64) {          // qkv = reps @ in_w.T + b
    int i = idx / kQC, j = idx % kQC;
    float acc = ca_in_b[j];
    for (int k = 0; k < 64; ++k) acc += sR[i * 64 + k] * ca_in_w[j * 64 + k];
    sQ[idx] = acc;
  }
  __syncthreads();
  if (t < 28) {                                          // t = head*7 + token
    int hh = t / 7, i = t % 7;
    float s[7], mx = -1e30f;
    for (int j = 0; j < 7; ++j) {
      float a = 0.f;
      for (int d2 = 0; d2 < 16; ++d2)
        a += sQ[i * kQC + hh * 16 + d2] * sQ[j * kQC + 64 + hh * 16 + d2];
      s[j] = a * 0.25f;
      mx = fmaxf(mx, s[j]);
    }
    float den = 0.f;
    for (int j = 0; j < 7; ++j) { s[j] = __expf(s[j] - mx); den += s[j]; }
    float inv = 1.f / den;
    for (int d2 = 0; d2 < 16; ++d2) {
      float o = 0.f;
      for (int j = 0; j < 7; ++j) o += s[j] * sQ[j * kQC + 128 + hh * 16 + d2];
      sO[i * 64 + hh * 16 + d2] = o * inv;
    }
  }
  if (t == 0) {                                          // fusion softmax
    float mx = fusion_w[0];
    for (int i = 1; i < 7; ++i) mx = fmaxf(mx, fusion_w[i]);
    float den = 0.f;
    for (int i = 0; i < 7; ++i) { sW[i] = __expf(fusion_w[i] - mx); den += sW[i]; }
    for (int i = 0; i < 7; ++i) sW[i] /= den;
  }
  __syncthreads();
  for (int idx = t; idx < 7 * 64; idx += 64) {           // cross = o @ out_w.T + b
    int i = idx / 64, j = idx % 64;
    float acc = ca_out_b[j];
    for (int k = 0; k < 64; ++k) acc += sO[i * 64 + k] * ca_out_w[j * 64 + k];
    sC[idx] = acc;
  }
  __syncthreads();
  if (t < 64) {                                          // fused = (cross*w) @ fc_w + b
    float acc = fc_b[t];
    for (int i = 0; i < 7; ++i) {
      float w = sW[i];
      for (int k = 0; k < 64; ++k)
        acc += sC[i * 64 + k] * w * fc_w[(i * 64 + k) * 64 + t];
    }
    sF[t] = acc;
  }
  __syncthreads();
  if (t < 32) {                                          // relu(fused @ c1 + b)
    float acc = c1_b[t];
    for (int k = 0; k < 64; ++k) acc += sF[k] * c1_w[k * 32 + t];
    sHb[t] = acc > 0.f ? acc : 0.f;
  }
  __syncthreads();
  if (t < 2) {                                           // @ c2 + b
    float acc = c2_b[t];
    for (int k = 0; k < 32; ++k) acc += sHb[k] * c2_w[k * 2 + t];
    outp[t] = acc;
  }
}

// ---------------------------------------------------------------------------
// Launch
// ---------------------------------------------------------------------------
extern "C" void kernel_launch(void* const* d_in, const int* in_sizes, int n_in,
                              void* d_out, int out_size, void* d_ws, size_t ws_size,
                              hipStream_t stream) {
  (void)in_sizes; (void)n_in; (void)out_size; (void)ws_size;
  const float* x        = (const float*)d_in[0];
  const int*   ei       = (const int*)  d_in[1];
  const float* g1_w     = (const float*)d_in[3];
  const float* g1_as    = (const float*)d_in[4];
  const float* g1_ad    = (const float*)d_in[5];
  const float* g1_b     = (const float*)d_in[6];
  const float* g2_w     = (const float*)d_in[7];
  const float* g2_as    = (const float*)d_in[8];
  const float* g2_ad    = (const float*)d_in[9];
  const float* g2_b     = (const float*)d_in[10];
  const float* sa_in_w  = (const float*)d_in[11];
  const float* sa_in_b  = (const float*)d_in[12];
  const float* sa_out_w = (const float*)d_in[13];
  const float* sa_out_b = (const float*)d_in[14];
  const float* ca_in_w  = (const float*)d_in[15];
  const float* ca_in_b  = (const float*)d_in[16];
  const float* ca_out_w = (const float*)d_in[17];
  const float* ca_out_b = (const float*)d_in[18];
  const float* fusion_w = (const float*)d_in[19];
  const float* fc_w     = (const float*)d_in[20];
  const float* fc_b     = (const float*)d_in[21];
  const float* c1_w     = (const float*)d_in[22];
  const float* c1_b     = (const float*)d_in[23];
  const float* c2_w     = (const float*)d_in[24];
  const float* c2_b     = (const float*)d_in[25];
  float* out = (float*)d_out;

  char* ws = (char*)d_ws;
  size_t off = 0;
  auto alloc = [&](size_t bytes) -> void* {
    off = (off + 255) & ~(size_t)255;
    void* p = ws + off; off += bytes; return p;
  };
  __bf16*   xb    = (__bf16*)  alloc((size_t)kN * kF * 2);
  __bf16*   w1pk  = (__bf16*)  alloc((size_t)kM * kF * kHC * 2);
  __bf16*   w2pk  = (__bf16*)  alloc((size_t)kM * kHC * kD * 2);
  __bf16*   inwpk = (__bf16*)  alloc((size_t)kM * kD * kQC * 2);
  __bf16*   outwpk= (__bf16*)  alloc((size_t)kM * kD * kD * 2);
  float*    h1    = (float*)   alloc((size_t)kN * kHC * 4);
  __bf16*   h1g   = (__bf16*)  alloc((size_t)kN * kHC * 2);   // bf16 gather shadow
  float*    out1  = (float*)   alloc((size_t)kN * kHC * 4);
  __bf16*   h1b   = (__bf16*)  alloc((size_t)kN * kHC * 2);
  float*    es1   = (float*)   alloc((size_t)kN * kH * 4);
  float*    ed1   = (float*)   alloc((size_t)kN * kH * 4);
  float*    ev1   = (float*)   alloc((size_t)kM * kET * kH * 4);
  unsigned* mx1   = (unsigned*)alloc((size_t)kN * kH * 4);
  float*    den1  = (float*)   alloc((size_t)kN * kH * 4);
  float*    h2    = (float*)   alloc((size_t)kN * kD * 4);
  __bf16*   h2g   = (__bf16*)  alloc((size_t)kN * kD * 2);    // bf16 gather shadow
  float*    out2  = (float*)   alloc((size_t)kN * kD * 4);
  __bf16*   h2b   = (__bf16*)  alloc((size_t)kN * kD * 2);
  float*    es2   = (float*)   alloc((size_t)kN * 4);
  float*    ed2   = (float*)   alloc((size_t)kN * 4);
  float*    ev2   = (float*)   alloc((size_t)kM * kET * 4);
  unsigned* mx2   = (unsigned*)alloc((size_t)kN * 4);
  float*    den2  = (float*)   alloc((size_t)kN * 4);
  float*    qkvf  = (float*)   alloc((size_t)kN * kQC * 4);
  __bf16*   qkvb  = (__bf16*)  alloc((size_t)kN * kQC * 2);
  __bf16*   vpk   = (__bf16*)  alloc((size_t)kM * kH * kNM * 16 * 2);
  float*    attn  = (float*)   alloc((size_t)kN * kD * 4);
  __bf16*   attnb = (__bf16*)  alloc((size_t)kN * kD * 2);
  float*    mha   = (float*)   alloc((size_t)kN * kD * 4);

  auto cdiv = [](int a, int b) { return (a + b - 1) / b; };

  // --- init accumulators (every call: deterministic work) -------------------
  const unsigned negInfEnc = 0x007FFFFFu; // fenc(-inf)
  k_fill_u32<<<cdiv(kN * kHC, 256), 256, 0, stream>>>((unsigned*)out1, 0u, kN * kHC);
  k_fill_u32<<<cdiv(kN * kD, 256), 256, 0, stream>>>((unsigned*)out2, 0u, kN * kD);
  k_fill_u32<<<cdiv(kN * kH, 256), 256, 0, stream>>>((unsigned*)den1, 0u, kN * kH);
  k_fill_u32<<<cdiv(kN, 256), 256, 0, stream>>>((unsigned*)den2, 0u, kN);
  k_fill_u32<<<cdiv(kN * kH, 256), 256, 0, stream>>>(mx1, negInfEnc, kN * kH);
  k_fill_u32<<<cdiv(kN, 256), 256, 0, stream>>>(mx2, negInfEnc, kN);

  // --- operand conversion / packing ----------------------------------------
  k_cvt_bf16_v4<<<cdiv(kN * kF / 4, 256), 256, 0, stream>>>(x, xb, kN * kF / 4);
  k_pack_b<<<cdiv(kM * kF * kHC, 256), 256, 0, stream>>>(g1_w, w1pk, kF, kHC, 0, kM * kF * kHC);
  k_pack_b<<<cdiv(kM * kHC * kD, 256), 256, 0, stream>>>(g2_w, w2pk, kHC, kD, 0, kM * kHC * kD);
  k_pack_b<<<cdiv(kM * kD * kQC, 256), 256, 0, stream>>>(sa_in_w, inwpk, kD, kQC, 1, kM * kD * kQC);
  k_pack_b<<<cdiv(kM * kD * kD, 256), 256, 0, stream>>>(sa_out_w, outwpk, kD, kD, 1, kM * kD * kD);

  auto gemm = [&](const __bf16* A, const __bf16* B, float* C, int Md, int Nd, int Kd,
                  int lda, int ldc, long long sA, long long sB, long long sC) {
    int tiles = (Md / 16) * (Nd / 16);
    dim3 grid(cdiv(tiles, 8), 1, kM);
    k_gemm_bf16<<<grid, 256, 0, stream>>>(A, B, C, Md, Nd, Kd, lda, ldc, sA, sB, sC);
  };

  // --- GAT layer 1 ----------------------------------------------------------
  gemm(xb, w1pk, h1, kNM, kHC, kF, kF, kHC,
       (long long)kNM * kF, (long long)kF * kHC, (long long)kNM * kHC);
  k_cvt_bf16_v4<<<cdiv(kN * kHC / 4, 256), 256, 0, stream>>>(h1, h1g, kN * kHC / 4);
  k_att_coef<<<cdiv(kN * kH, 128), 128, 0, stream>>>(h1, g1_as, g1_ad, es1, ed1, kH, kD);
  k_edge_p1<<<cdiv(kM * kET * kH, 256), 256, 0, stream>>>(ei, es1, ed1, ev1, mx1, kH);
  k_edge_p2<<<cdiv(kM * kET * kH, 256), 256, 0, stream>>>(ei, ev1, mx1, den1, kH);
  k_edge_p3<<<cdiv(kM * kET * kHC, 256), 256, 0, stream>>>(ei, ev1, den1, h1g, out1, kH, kD);
  k_bias_relu_cvt<<<cdiv(kN * kHC, 256), 256, 0, stream>>>(out1, g1_b, h1b, kHC);

  // --- GAT layer 2 ----------------------------------------------------------
  gemm(h1b, w2pk, h2, kNM, kD, kHC, kHC, kD,
       (long long)kNM * kHC, (long long)kHC * kD, (long long)kNM * kD);
  k_cvt_bf16_v4<<<cdiv(kN * kD / 4, 256), 256, 0, stream>>>(h2, h2g, kN * kD / 4);
  k_att_coef<<<cdiv(kN, 128), 128, 0, stream>>>(h2, g2_as, g2_ad, es2, ed2, 1, kD);
  k_edge_p1<<<cdiv(kM * kET, 256), 256, 0, stream>>>(ei, es2, ed2, ev2, mx2, 1);
  k_edge_p2<<<cdiv(kM * kET, 256), 256, 0, stream>>>(ei, ev2, mx2, den2, 1);
  k_edge_p3<<<cdiv(kM * kET * kD, 256), 256, 0, stream>>>(ei, ev2, den2, h2g, out2, 1, kD);
  k_bias_relu_cvt<<<cdiv(kN * kD, 256), 256, 0, stream>>>(out2, g2_b, h2b, kD);

  // --- module self-attention ------------------------------------------------
  gemm(h2b, inwpk, qkvf, kNM, kQC, kD, kD, kQC,
       (long long)kNM * kD, (long long)kD * kQC, (long long)kNM * kQC);
  k_bias_cvt<<<cdiv(kN * kQC, 256), 256, 0, stream>>>(qkvf, sa_in_b, qkvb, kQC);
  k_pack_v<<<cdiv(kM * kH * kNM * 16, 256), 256, 0, stream>>>(qkvf, sa_in_b, vpk);
  dim3 ag(kNM / 16, kH, kM);
  k_attn<<<ag, 256, 0, stream>>>(qkvb, vpk, attn);
  k_cvt_bf16_v4<<<cdiv(kN * kD / 4, 256), 256, 0, stream>>>(attn, attnb, kN * kD / 4);
  gemm(attnb, outwpk, mha, kNM, kD, kD, kD, kD,
       (long long)kNM * kD, (long long)kD * kD, (long long)kNM * kD);
  k_rowmean<<<kM, 256, 0, stream>>>(mha, sa_out_b, out + 2);   // reps -> d_out[2..449]

  // --- cross-attention + head ----------------------------------------------
  k_head<<<1, 64, 0, stream>>>(out + 2, ca_in_w, ca_in_b, ca_out_w, ca_out_b,
                               fusion_w, fc_w, fc_b, c1_w, c1_b, c2_w, c2_b, out);
}
